// TransDecoder_13426067767623
// MI455X (gfx1250) — matrix operand order repaired and checked by reading, not compiled, not run on hardware
//
#include <hip/hip_runtime.h>
#include <hip/hip_bf16.h>

// ---------------------------------------------------------------------------
// CDNA5 (gfx1250) transformer block: WMMA f32<=f16 for all GEMMs, bias folded
// into a 49-bucket per-query table (avoids the 655MB materialized bias), exact
// top-k(32) via wave32 iterative argmax in LDS, softmax + attn*V via WMMA.
// ---------------------------------------------------------------------------

typedef __attribute__((ext_vector_type(16))) _Float16     v16h;
typedef __attribute__((ext_vector_type(8)))  float        v8f;
typedef __attribute__((ext_vector_type(4)))  unsigned int u32x4;

union FragH {
  u32x4    u[2];
  v16h     h;
  _Float16 f[16];
};

constexpr int SEQ    = 1600;   // 40*40
constexpr int DIMN   = 384;
constexpr int NHEAD  = 6;
constexpr int DHEAD  = 64;
constexpr int INNERN = 384;    // NHEAD*DHEAD
constexpr int HIDN   = 1536;
constexpr int GW     = 40;
constexpr int NQB    = 100;    // SEQ/16
constexpr int LAYERS = 6;
constexpr float SCALE = 0.05103103630798288f;  // 384^-0.5
constexpr float NEGMAX = -3.4028234663852886e38f;

// Attention dynamic LDS: scores[16][1600] f32 + qbias[16][49] f32
constexpr unsigned ATTN_SMEM = (16 * SEQ + 16 * 49) * sizeof(float);

// ---- fragment loaders -----------------------------------------------------
// A (16x32 f16): lane m=lane%16; lanes<16: K 0-7 & 16-23, lanes>=16: 8-15 & 24-31
__device__ __forceinline__ void load_fragA(FragH& f, const _Float16* rowp, int kk, int kb) {
  const char* p = (const char*)rowp + 2 * (kk + kb);
  f.u[0] = *(const u32x4*)(p);
  f.u[1] = *(const u32x4*)(p + 32);
}
// B (32x16 f16) from BT[N][K] row-major: lane n=lane%16; lanes<16: K 0-7 & 16-23, >=16: 8-15 & 24-31
__device__ __forceinline__ void load_fragB(FragH& f, const _Float16* rowp, int kk, int kb) {
  const char* p = (const char*)rowp + 2 * (kk + kb);
  f.u[0] = *(const u32x4*)(p);
  f.u[1] = *(const u32x4*)(p + 32);
}
__device__ __forceinline__ v8f wmma_f16(const FragH& a, const FragH& b, v8f c) {
  return __builtin_amdgcn_wmma_f32_16x16x32_f16(false, a.h, false, b.h, (short)0, c,
                                                false, false);
}

// ---- weight transpose + f32->f16 convert: out[n*K+k] = in[k*N+n] ----------
__global__ void transpose_cvt_kernel(const float* __restrict__ in, _Float16* __restrict__ out,
                                     int K, int N) {
  int i = blockIdx.x * 256 + threadIdx.x;
  if (i < K * N) {
    int k = i / N, n = i % N;
    out[(size_t)n * K + k] = (_Float16)in[i];
  }
}

// ---- layernorm over DIMN=384, f16 output ----------------------------------
__launch_bounds__(128)
__global__ void ln16_kernel(const float* __restrict__ h, const float* __restrict__ g,
                            const float* __restrict__ b, _Float16* __restrict__ out) {
  __shared__ float red[128];
  const int s = blockIdx.x, t = threadIdx.x;
  const float* row = h + (size_t)s * DIMN;
  float x0 = row[t], x1 = row[t + 128], x2 = row[t + 256];
  red[t] = x0 + x1 + x2;
  __syncthreads();
  for (int off = 64; off > 0; off >>= 1) { if (t < off) red[t] += red[t + off]; __syncthreads(); }
  float mean = red[0] * (1.f / DIMN);
  __syncthreads();
  float d0 = x0 - mean, d1 = x1 - mean, d2 = x2 - mean;
  red[t] = d0 * d0 + d1 * d1 + d2 * d2;
  __syncthreads();
  for (int off = 64; off > 0; off >>= 1) { if (t < off) red[t] += red[t + off]; __syncthreads(); }
  float inv = rsqrtf(red[0] * (1.f / DIMN) + 1e-5f);
  out[(size_t)s * DIMN + t      ] = (_Float16)(d0 * inv * g[t      ] + b[t      ]);
  out[(size_t)s * DIMN + t + 128] = (_Float16)(d1 * inv * g[t + 128] + b[t + 128]);
  out[(size_t)s * DIMN + t + 256] = (_Float16)(d2 * inv * g[t + 256] + b[t + 256]);
}

// ---- generic WMMA GEMM: C[M,N] = A[M,K](f16) * BT[N,K](f16) ---------------
// MODE 0: QKV stride-3 de-interleave scatter to q16/k16/vT16 (f16)
// MODE 1: + bias, leaky_relu(0.2), f16 out
// MODE 2: residual accumulate into f32 C (+ optional bias)
template <int MODE>
__launch_bounds__(256)
__global__ void gemm16_kernel(const _Float16* __restrict__ A, const _Float16* __restrict__ BT,
                              int M, int N, int K,
                              float* __restrict__ Cres, const float* __restrict__ bias,
                              _Float16* __restrict__ O16,
                              _Float16* __restrict__ Q16, _Float16* __restrict__ K16,
                              _Float16* __restrict__ VT16) {
  const int lane = threadIdx.x & 31, wid = threadIdx.x >> 5;
  const int m0 = blockIdx.y * 32 + (wid >> 2) * 16;
  const int n0 = blockIdx.x * 64 + (wid & 3) * 16;
  const int kbA = (lane < 16) ? 0 : 8;
  const int kbB = (lane < 16) ? 0 : 8;
  const _Float16* arow = A  + (size_t)(m0 + (lane & 15)) * K;
  const _Float16* brow = BT + (size_t)(n0 + (lane & 15)) * K;
  v8f acc = {};
  for (int kk = 0; kk < K; kk += 32) {
    if (kk + 32 < K) {
      __builtin_prefetch(arow + kk + 32, 0, 0);
      __builtin_prefetch(brow + kk + 32, 0, 0);
    }
    FragH a, b;
    load_fragA(a, arow, kk, kbA);
    load_fragB(b, brow, kk, kbB);
    acc = wmma_f16(a, b, acc);
  }
  // C layout: VGPR e -> M = e + (lane<16?0:8), N = lane%16
  const int nn = n0 + (lane & 15);
  const int mbase = m0 + ((lane < 16) ? 0 : 8);
#pragma unroll
  for (int e = 0; e < 8; ++e) {
    const int mm = mbase + e;
    const float v = acc[e];
    if (MODE == 0) {
      // qkv column nn: t=nn%3 (q/k/v), idx=nn/3 -> head=idx/64, d=idx%64
      const int t = nn % 3, idx = nn / 3, hh = idx >> 6, dd = idx & 63;
      const _Float16 hv = (_Float16)v;
      if (t == 0)      Q16[((size_t)hh * SEQ + mm) * DHEAD + dd] = hv;
      else if (t == 1) K16[((size_t)hh * SEQ + mm) * DHEAD + dd] = hv;
      else             VT16[((size_t)hh * DHEAD + dd) * SEQ + mm] = hv;
    } else if (MODE == 1) {
      float x = v + bias[nn];
      x = x > 0.f ? x : 0.2f * x;
      O16[(size_t)mm * N + nn] = (_Float16)x;
    } else {
      const float x = v + (bias ? bias[nn] : 0.f);
      Cres[(size_t)mm * N + nn] += x;
    }
  }
}

// ---- attention: one WG (8 wave32) per (head, 16-query tile) ---------------
__launch_bounds__(256)
__global__ void attn_kernel(const _Float16* __restrict__ Q16, const _Float16* __restrict__ K16,
                            const _Float16* __restrict__ VT16, const float* __restrict__ rel_bias,
                            _Float16* __restrict__ O16) {
  const int head = blockIdx.x / NQB;
  const int q0 = (blockIdx.x % NQB) * 16;
  const int tid = threadIdx.x, lane = tid & 31, wid = tid >> 5;
  extern __shared__ float smem_f[];
  float* sc = smem_f;             // [16][SEQ] scores -> probs
  float* qb = sc + 16 * SEQ;      // [16][49]  per-query bucket dots

  // per-query bias table: qb[r][nb] = q[r,:] . rel_bias[nb,:]
  for (int t = tid; t < 16 * 49; t += 256) {
    const int r = t / 49, nb = t % 49;
    const _Float16* qr = Q16 + ((size_t)head * SEQ + q0 + r) * DHEAD;
    const float* br = rel_bias + nb * DHEAD;
    float s = 0.f;
    for (int d = 0; d < DHEAD; ++d) s += (float)qr[d] * br[d];
    qb[t] = s;
  }

  const int kbA = (lane < 16) ? 0 : 8;
  const int kbB = (lane < 16) ? 0 : 8;
  const _Float16* qrow = Q16 + ((size_t)head * SEQ + q0 + (lane & 15)) * DHEAD;
  FragH a_lo, a_hi;
  load_fragA(a_lo, qrow, 0, kbA);
  load_fragA(a_hi, qrow, 32, kbA);
  __syncthreads();

  // scores (q.k^T + q.bias) * scale, 16x16 tiles, waves split the 100 k-blocks
  for (int kb2 = wid; kb2 < NQB; kb2 += 8) {
    const int k0 = kb2 * 16;
    const _Float16* krow = K16 + ((size_t)head * SEQ + k0 + (lane & 15)) * DHEAD;
    FragH b_lo, b_hi;
    load_fragB(b_lo, krow, 0, kbB);
    load_fragB(b_hi, krow, 32, kbB);
    v8f c = {};
    c = wmma_f16(a_lo, b_lo, c);
    c = wmma_f16(a_hi, b_hi, c);
    const int nn = lane & 15, kpos = k0 + nn;
    const int ky = kpos / GW, kx = kpos % GW;
    const int mbase = (lane < 16) ? 0 : 8;
#pragma unroll
    for (int e = 0; e < 8; ++e) {
      const int m = mbase + e, qpos = q0 + m;
      const int dv = ky - qpos / GW, dh = kx - qpos % GW;
      const int adv = dv < 0 ? -dv : dv, adh = dh < 0 ? -dh : dh;
      const int bkt = (adv + adh <= 3) ? (dv + 3) * 7 + (dh + 3) : 0;
      sc[m * SEQ + kpos] = (c[e] + qb[m * 49 + bkt]) * SCALE;
    }
  }
  __syncthreads();

  // exact top-32 threshold per row (wave argmax x32, destructive w/ restore),
  // then masked softmax in place. Wave w owns rows w and w+8.
  for (int rr = 0; rr < 2; ++rr) {
    const int r = wid + rr * 8;
    float* row = sc + r * SEQ;
    float thres = 0.f, rowmax = 0.f, saveV = 0.f;
    int saveI = 0;
    for (int it = 0; it < 32; ++it) {
      float mv = NEGMAX;
      int mi = 0;
      for (int j = lane; j < SEQ; j += 32) {
        const float v = row[j];
        if (v > mv) { mv = v; mi = j; }
      }
      for (int off = 16; off > 0; off >>= 1) {
        const float ov = __shfl_xor(mv, off, 32);
        const int oi = __shfl_xor(mi, off, 32);
        if (ov > mv || (ov == mv && oi < mi)) { mv = ov; mi = oi; }
      }
      if (it == 0) rowmax = mv;
      thres = mv;
      if (lane == it) { saveV = mv; saveI = mi; }   // lane L archives iteration L
      if (lane == 0) row[mi] = NEGMAX;              // evict for next scan
    }
    row[saveI] = saveV;  // restore originals (32 distinct indices, one per lane)
    float sum = 0.f;
    for (int j = lane; j < SEQ; j += 32) {
      const float v = row[j];
      const float p = (v >= thres) ? __expf(v - rowmax) : 0.f;
      row[j] = p;
      sum += p;
    }
    for (int off = 16; off > 0; off >>= 1) sum += __shfl_xor(sum, off, 32);
    const float inv = 1.f / sum;
    for (int j = lane; j < SEQ; j += 32) row[j] *= inv;
  }
  __syncthreads();

  // attn * V: waves 0..3 each own a 16-wide d-tile, reduce over all 1600 keys
  if (wid < 4) {
    const int d0 = wid * 16;
    const _Float16* vrow = VT16 + ((size_t)head * DHEAD + d0 + (lane & 15)) * SEQ;
    const int m = lane & 15;
    v8f acc = {};
    for (int kk = 0; kk < SEQ; kk += 32) {
      FragH a, b;
#pragma unroll
      for (int rg = 0; rg < 8; ++rg) {  // build A frag from f32 probs in LDS
        const int k = kk + ((rg < 4) ? (kbA + 2 * rg) : (kbA + 16 + 2 * (rg - 4)));
        a.f[2 * rg]     = (_Float16)sc[m * SEQ + k];
        a.f[2 * rg + 1] = (_Float16)sc[m * SEQ + k + 1];
      }
      load_fragB(b, vrow, kk, kbB);
      acc = wmma_f16(a, b, acc);
    }
    const int nn = lane & 15, mbase = (lane < 16) ? 0 : 8;
#pragma unroll
    for (int e = 0; e < 8; ++e)
      O16[(size_t)(q0 + mbase + e) * INNERN + head * DHEAD + d0 + nn] = (_Float16)acc[e];
  }
}

// ---------------------------------------------------------------------------
extern "C" void kernel_launch(void* const* d_in, const int* in_sizes, int n_in,
                              void* d_out, int out_size, void* d_ws, size_t ws_size,
                              hipStream_t stream) {
  (void)in_sizes; (void)n_in; (void)out_size; (void)ws_size;
  const float* x      = (const float*)d_in[0];
  const float* W_qkv  = (const float*)d_in[1];
  const float* W_o    = (const float*)d_in[2];
  const float* ln1_g  = (const float*)d_in[3];
  const float* ln1_b  = (const float*)d_in[4];
  const float* ln2_g  = (const float*)d_in[5];
  const float* ln2_b  = (const float*)d_in[6];
  const float* relb   = (const float*)d_in[7];
  const float* ff_w1  = (const float*)d_in[8];
  const float* ff_b1  = (const float*)d_in[9];
  const float* ff_w2  = (const float*)d_in[10];
  const float* ff_b2  = (const float*)d_in[11];
  // d_in[12..14] = H=40, W=40, topk=32 (hardcoded per setup_inputs)

  char* ws = (char*)d_ws;
  size_t off = 0;
  auto take = [&](size_t bytes) -> void* {
    void* r = ws + off;
    off = (off + bytes + 255) & ~(size_t)255;
    return r;
  };
  float*     hbuf  = (float*)take((size_t)SEQ * DIMN * 4);
  _Float16*  n16   = (_Float16*)take((size_t)SEQ * DIMN * 2);
  _Float16*  q16   = (_Float16*)take((size_t)NHEAD * SEQ * DHEAD * 2);
  _Float16*  k16   = (_Float16*)take((size_t)NHEAD * SEQ * DHEAD * 2);
  _Float16*  vT16  = (_Float16*)take((size_t)NHEAD * SEQ * DHEAD * 2);
  _Float16*  o16   = (_Float16*)take((size_t)SEQ * INNERN * 2);
  _Float16*  a16   = (_Float16*)take((size_t)SEQ * HIDN * 2);
  _Float16*  WqkvT = (_Float16*)take((size_t)LAYERS * 1152 * 384 * 2);
  _Float16*  WoT   = (_Float16*)take((size_t)LAYERS * 384 * 384 * 2);
  _Float16*  W1T   = (_Float16*)take((size_t)LAYERS * 1536 * 384 * 2);
  _Float16*  W2T   = (_Float16*)take((size_t)LAYERS * 384 * 1536 * 2);

  hipFuncSetAttribute(reinterpret_cast<const void*>(attn_kernel),
                      hipFuncAttributeMaxDynamicSharedMemorySize, (int)ATTN_SMEM);

  hipMemcpyAsync(hbuf, x, (size_t)SEQ * DIMN * 4, hipMemcpyDeviceToDevice, stream);

  // weight transpose + f16 convert (once per launch; all layers)
  for (int l = 0; l < LAYERS; ++l) {
    int n;
    n = 384 * 1152;
    transpose_cvt_kernel<<<(n + 255) / 256, 256, 0, stream>>>(W_qkv + (size_t)l * n, WqkvT + (size_t)l * n, 384, 1152);
    n = 384 * 384;
    transpose_cvt_kernel<<<(n + 255) / 256, 256, 0, stream>>>(W_o + (size_t)l * n, WoT + (size_t)l * n, 384, 384);
    n = 384 * 1536;
    transpose_cvt_kernel<<<(n + 255) / 256, 256, 0, stream>>>(ff_w1 + (size_t)l * n, W1T + (size_t)l * n, 384, 1536);
    n = 1536 * 384;
    transpose_cvt_kernel<<<(n + 255) / 256, 256, 0, stream>>>(ff_w2 + (size_t)l * n, W2T + (size_t)l * n, 1536, 384);
  }

  for (int l = 0; l < LAYERS; ++l) {
    ln16_kernel<<<SEQ, 128, 0, stream>>>(hbuf, ln1_g + l * DIMN, ln1_b + l * DIMN, n16);
    gemm16_kernel<0><<<dim3(18, 50), 256, 0, stream>>>(
        n16, WqkvT + (size_t)l * 1152 * 384, SEQ, 1152, 384,
        nullptr, nullptr, nullptr, q16, k16, vT16);
    attn_kernel<<<NHEAD * NQB, 256, ATTN_SMEM, stream>>>(q16, k16, vT16, relb, o16);
    gemm16_kernel<2><<<dim3(6, 50), 256, 0, stream>>>(
        o16, WoT + (size_t)l * 384 * 384, SEQ, 384, 384,
        hbuf, nullptr, nullptr, nullptr, nullptr, nullptr);
    ln16_kernel<<<SEQ, 128, 0, stream>>>(hbuf, ln2_g + l * DIMN, ln2_b + l * DIMN, n16);
    gemm16_kernel<1><<<dim3(24, 50), 256, 0, stream>>>(
        n16, W1T + (size_t)l * 1536 * 384, SEQ, 1536, 384,
        nullptr, ff_b1 + (size_t)l * HIDN, a16, nullptr, nullptr, nullptr);
    gemm16_kernel<2><<<dim3(6, 50), 256, 0, stream>>>(
        a16, W2T + (size_t)l * 384 * 1536, SEQ, 384, 1536,
        hbuf, ff_b2 + (size_t)l * DIMN, nullptr, nullptr, nullptr, nullptr);
  }

  hipMemcpyAsync(d_out, hbuf, (size_t)SEQ * DIMN * 4, hipMemcpyDeviceToDevice, stream);
}
